// NexToU_Encoder_17042430231091
// MI455X (gfx1250) — matrix operand order repaired and checked by reading, not compiled, not run on hardware
//
#include <hip/hip_runtime.h>
#include <hip/hip_bf16.h>

typedef __attribute__((ext_vector_type(16))) __bf16 v16bf;
typedef __attribute__((ext_vector_type(8)))  __bf16 v8bf;
typedef __attribute__((ext_vector_type(8)))  float  v8f;

#define WB 2
#define WC 128
#define WN 4096
#define KNN 9
#define CHUNK 512

// ---------- small staging kernels ----------

__global__ void cvt_f32_bf16(const float* __restrict__ s, __bf16* __restrict__ d, int n) {
  int i = blockIdx.x * blockDim.x + threadIdx.x;
  if (i < n) d[i] = (__bf16)s[i];
}

// (B, Cc, N) f32 -> (B, N, Cc) bf16
__global__ void transpose_cvt(const float* __restrict__ s, __bf16* __restrict__ d,
                              int Cc, int N, int B) {
  int i = blockIdx.x * blockDim.x + threadIdx.x;
  int total = B * Cc * N;
  if (i >= total) return;
  int n = i % N;
  int c = (i / N) % Cc;
  int b = i / (N * Cc);
  d[((size_t)b * N + n) * Cc + c] = (__bf16)s[i];
}

// ---------- bf16 WMMA GEMM:  out[b,m,n] = sum_k A[m,k] * Bt[b,n,k] + bias[m] ----------
// A: M x K row-major bf16 ; Bt: B x N x K bf16 ; out: B x M x N f32.
// grid(N/64, M/16, B), block 128 (4 waves, each wave one 16x16 tile along N).
// The 16xK A tile is shared by all 4 waves -> staged ONCE into LDS with the
// gfx1250 async-copy engine (global_load_async_to_lds_b128 / ASYNCcnt), then
// WMMA A-fragments are fed from LDS; B streams straight from global (no
// intra-block reuse, L2 handles cross-block reuse).

__global__ __launch_bounds__(128) void gemm_bf16_wmma(
    const __bf16* __restrict__ A, const __bf16* __restrict__ Bt,
    const float* __restrict__ bias, float* __restrict__ out,
    int M, int N, int K) {
  __shared__ __bf16 aTile[16 * 512];   // 16 KB, supports K up to 512

  int tid  = threadIdx.x;
  int lane = tid & 31;
  int wave = tid >> 5;
  int tn = blockIdx.x * 4 + wave;
  int tm = blockIdx.y;
  int b  = blockIdx.z;
  int m  = lane & 15;          // A row within tile
  int kh = lane >> 4;          // K-half selector (also C/D row-half selector)
  int col = lane & 15;         // B/D column within tile

  // ---- async-stage the 16xK weight tile into LDS (2 KB per block-wide issue) ----
  {
    unsigned lds_base = (unsigned)(size_t)(&aTile[0]);   // wave-relative LDS offset
    size_t gbase = (size_t)(A + (size_t)tm * 16 * K);
    int iters = K >> 6;                                  // (16*K*2) / (128*16)
    for (int it = 0; it < iters; ++it) {
      unsigned byte_off = (unsigned)((it * 128 + tid) * 16);
      unsigned ldsoff = lds_base + byte_off;
      unsigned long long gaddr = (unsigned long long)(gbase + byte_off);
      asm volatile("global_load_async_to_lds_b128 %0, %1, off"
                   :: "v"(ldsoff), "v"(gaddr) : "memory");
    }
    asm volatile("s_wait_asynccnt 0" ::: "memory");
  }
  __syncthreads();

  const __bf16* brow = Bt + ((size_t)b * N + tn * 16 + col) * K;

  v8f c = {};
  for (int kk = 0; kk < K; kk += 32) {
    // A fragment from LDS: lanes 0-15 hold K=[kk..+8)&[kk+16..+24); lanes 16-31 the +8 chunks
    v8bf alo = *(const v8bf*)(aTile + m * K + kk + kh * 8);
    v8bf ahi = *(const v8bf*)(aTile + m * K + kk + 16 + kh * 8);
    v16bf a;
#pragma unroll
    for (int i = 0; i < 8; ++i) { a[i] = alo[i]; a[i + 8] = ahi[i]; }
    // B fragment: lane holds one column's K=[kk+kh*8..+8)&[kk+16+kh*8..+8)
    v8bf blo = *(const v8bf*)(brow + kk + kh * 8);
    v8bf bhi = *(const v8bf*)(brow + kk + 16 + kh * 8);
    v16bf bf;
#pragma unroll
    for (int i = 0; i < 8; ++i) { bf[i] = blo[i]; bf[i + 8] = bhi[i]; }
    if (kk + 32 < K) __builtin_prefetch(brow + kk + 32, 0, 0);  // global_prefetch_b8
    c = __builtin_amdgcn_wmma_f32_16x16x32_bf16(false, a, false, bf,
                                                (short)0, c, false, false);
  }
#pragma unroll
  for (int r = 0; r < 8; ++r) {
    int row = tm * 16 + r + kh * 8;   // lanes 0-15 -> M 0..7, lanes 16-31 -> M 8..15
    out[((size_t)b * M + row) * N + tn * 16 + col] = c[r] + bias[row];
  }
}

// ---------- fused BatchNorm (training stats) + optional relu/residual/bf16T output ----------
// y: B x Cc x N f32, normalized in place. grid(Cc), block 256.

__global__ __launch_bounds__(256) void bn_fused(
    float* __restrict__ y, const float* __restrict__ gamma,
    const float* __restrict__ beta, const float* __restrict__ resid,
    __bf16* __restrict__ outT, int Cc, int N, int B, int relu) {
  __shared__ float r0[256], r1[256];
  int c = blockIdx.x, tid = threadIdx.x;
  int total = B * N;
  float s = 0.f, ss = 0.f;
  for (int i = tid; i < total; i += 256) {
    int b = i / N, n = i % N;
    float v = y[((size_t)b * Cc + c) * N + n];
    s += v; ss += v * v;
  }
  r0[tid] = s; r1[tid] = ss;
  __syncthreads();
  for (int o = 128; o > 0; o >>= 1) {
    if (tid < o) { r0[tid] += r0[tid + o]; r1[tid] += r1[tid + o]; }
    __syncthreads();
  }
  float mean = r0[0] / (float)total;
  float var  = r1[0] / (float)total - mean * mean;
  float inv  = rsqrtf(var + 1e-5f);
  float g = gamma[c], bb = beta[c];
  for (int i = tid; i < total; i += 256) {
    int b = i / N, n = i % N;
    size_t off = ((size_t)b * Cc + c) * N + n;
    float v = (y[off] - mean) * inv * g + bb;
    if (relu) v = fmaxf(v, 0.f);
    if (resid) v += resid[off];
    y[off] = v;
    if (outT) outT[((size_t)b * N + n) * Cc + c] = (__bf16)v;
  }
}

// ---------- per-node L2 normalize: (B,C,N) f32 -> (B,N,C) bf16 ----------

__global__ void normalize_nodes(const float* __restrict__ f, __bf16* __restrict__ fnT,
                                int Cc, int N, int B) {
  int i = blockIdx.x * blockDim.x + threadIdx.x;
  if (i >= B * N) return;
  int b = i / N, n = i % N;
  const float* p = f + (size_t)b * Cc * N + n;
  float ss = 0.f;
  for (int c = 0; c < Cc; ++c) { float v = p[(size_t)c * N]; ss += v * v; }
  float r = rsqrtf(ss + 1e-12f);
  __bf16* o = fnT + ((size_t)b * N + n) * Cc;
  for (int c = 0; c < Cc; ++c) o[c] = (__bf16)(p[(size_t)c * N] * r);
}

// ---------- fused kNN: WMMA Gram-matrix tiles + in-LDS top-9 ----------
// fnT: B x N x 128 bf16 (unit-norm rows) -> idx_out: B x N x 9 int.
// One block = 16 rows; 4 waves compute 16x512 inner-product chunks via WMMA,
// 128 threads keep per-(row,segment) top-9 in registers, merged at the end.
// (Rows are unit-norm, so top-k of 2*inner-2 == top-k of inner.)

__global__ __launch_bounds__(128) void knn_topk(
    const __bf16* __restrict__ fnT, int* __restrict__ idx_out, int N) {
  __shared__ float dist[16][CHUNK];          // 32 KB
  __shared__ float mbv[16][8][KNN];
  __shared__ int   mbi[16][8][KNN];
  const int Cc = 128;
  int b = blockIdx.y, rt = blockIdx.x;
  int tid = threadIdx.x, lane = tid & 31, wave = tid >> 5;
  int m = lane & 15, kh = lane >> 4;

  // Preload all 4 A K-fragments for this 16-row tile (rows = nodes rt*16..+15)
  const __bf16* arow = fnT + ((size_t)b * N + rt * 16 + m) * Cc;
  v16bf afr[4];
#pragma unroll
  for (int k4 = 0; k4 < 4; ++k4) {
    v8bf lo = *(const v8bf*)(arow + k4 * 32 + kh * 8);
    v8bf hi = *(const v8bf*)(arow + k4 * 32 + 16 + kh * 8);
#pragma unroll
    for (int i = 0; i < 8; ++i) { afr[k4][i] = lo[i]; afr[k4][i + 8] = hi[i]; }
  }

  float bv[KNN]; int bi[KNN];
#pragma unroll
  for (int s = 0; s < KNN; ++s) { bv[s] = -3.0e38f; bi[s] = 0; }
  int row = tid & 15, seg = tid >> 4;

  for (int ch = 0; ch < N / CHUNK; ++ch) {
    for (int t = 0; t < 8; ++t) {          // each wave: 8 of 32 column tiles
      int ct = ch * (CHUNK / 16) + wave * 8 + t;
      const __bf16* brow = fnT + ((size_t)b * N + ct * 16 + (lane & 15)) * Cc;
      v8f c = {};
#pragma unroll
      for (int k4 = 0; k4 < 4; ++k4) {
        v8bf blo = *(const v8bf*)(brow + k4 * 32 + kh * 8);
        v8bf bhi = *(const v8bf*)(brow + k4 * 32 + 16 + kh * 8);
        v16bf bf;
#pragma unroll
        for (int i = 0; i < 8; ++i) { bf[i] = blo[i]; bf[i + 8] = bhi[i]; }
        c = __builtin_amdgcn_wmma_f32_16x16x32_bf16(false, afr[k4], false, bf,
                                                    (short)0, c, false, false);
      }
      int col = (wave * 8 + t) * 16 + (lane & 15);
#pragma unroll
      for (int r = 0; r < 8; ++r) dist[kh * 8 + r][col] = c[r];
    }
    __syncthreads();
    // top-9 update: thread (row, seg) scans 64 columns of this chunk
    for (int j = 0; j < 64; ++j) {
      int cc = seg * 64 + j;
      float v = dist[row][cc];
      int gi = ch * CHUNK + cc;
      if (v > bv[KNN - 1]) {
        bv[KNN - 1] = v; bi[KNN - 1] = gi;
#pragma unroll
        for (int s = KNN - 1; s > 0; --s)
          if (bv[s] > bv[s - 1]) {
            float tv = bv[s]; bv[s] = bv[s - 1]; bv[s - 1] = tv;
            int ti = bi[s]; bi[s] = bi[s - 1]; bi[s - 1] = ti;
          }
      }
    }
    __syncthreads();
  }

#pragma unroll
  for (int s = 0; s < KNN; ++s) { mbv[row][seg][s] = bv[s]; mbi[row][seg][s] = bi[s]; }
  __syncthreads();
  if (tid < 16) {
    float fv[KNN]; int fi[KNN];
#pragma unroll
    for (int s = 0; s < KNN; ++s) { fv[s] = -3.0e38f; fi[s] = 0; }
    for (int sg = 0; sg < 8; ++sg) {
#pragma unroll
      for (int s = 0; s < KNN; ++s) {
        float v = mbv[tid][sg][s]; int gi = mbi[tid][sg][s];
        if (v > fv[KNN - 1]) {
          fv[KNN - 1] = v; fi[KNN - 1] = gi;
#pragma unroll
          for (int q = KNN - 1; q > 0; --q)
            if (fv[q] > fv[q - 1]) {
              float tv = fv[q]; fv[q] = fv[q - 1]; fv[q - 1] = tv;
              int ti = fi[q]; fi[q] = fi[q - 1]; fi[q - 1] = ti;
            }
        }
      }
    }
    int node = rt * 16 + tid;
#pragma unroll
    for (int s = 0; s < KNN; ++s)
      idx_out[((size_t)b * N + node) * KNN + s] = fi[s];
  }
}

// ---------- gather neighbors + max-relative, write interleaved (B,N,2C) bf16 staging ----------

__global__ void gather_maxrel(const float* __restrict__ f, const int* __restrict__ idx,
                              __bf16* __restrict__ zT, int Cc, int N, int B) {
  int i = blockIdx.x * blockDim.x + threadIdx.x;
  int total = B * Cc * N;
  if (i >= total) return;
  int n = i % N;
  int c = (i / N) % Cc;
  int b = i / (N * Cc);
  const float* fc = f + ((size_t)b * Cc + c) * N;
  const int* ip = idx + ((size_t)b * N + n) * KNN;
  float fv = fc[n];
  float mx = -3.0e38f;
#pragma unroll
  for (int k = 0; k < KNN; ++k) mx = fmaxf(mx, fc[ip[k]]);
  float mr = mx - fv;
  __bf16* o = zT + ((size_t)b * N + n) * (2 * Cc);
  o[2 * c]     = (__bf16)fv;   // even channels: x
  o[2 * c + 1] = (__bf16)mr;   // odd channels: max-relative
}

// ---------- driver ----------

extern "C" void kernel_launch(void* const* d_in, const int* in_sizes, int n_in,
                              void* d_out, int out_size, void* d_ws, size_t ws_size,
                              hipStream_t stream) {
  const int B = WB, C = WC, N = WN;
  const float* x      = (const float*)d_in[0];
  const float* fc1_w  = (const float*)d_in[1];
  const float* fc1_b  = (const float*)d_in[2];
  const float* bn1_g  = (const float*)d_in[3];
  const float* bn1_b  = (const float*)d_in[4];
  const float* g_w    = (const float*)d_in[5];
  const float* g_b    = (const float*)d_in[6];
  const float* bng_g  = (const float*)d_in[7];
  const float* bng_b  = (const float*)d_in[8];
  const float* fc2_w  = (const float*)d_in[9];
  const float* fc2_b  = (const float*)d_in[10];
  const float* bn2_g  = (const float*)d_in[11];
  const float* bn2_b  = (const float*)d_in[12];
  const float* f1_w   = (const float*)d_in[13];
  const float* f1_b   = (const float*)d_in[14];
  const float* fbn1_g = (const float*)d_in[15];
  const float* fbn1_b = (const float*)d_in[16];
  const float* f2_w   = (const float*)d_in[17];
  const float* f2_b   = (const float*)d_in[18];
  const float* fbn2_g = (const float*)d_in[19];
  const float* fbn2_b = (const float*)d_in[20];
  float* out = (float*)d_out;

  // workspace carve-out
  char* ws = (char*)d_ws;
  size_t off = 0;
  auto carve = [&](size_t bytes) { char* p = ws + off; off = (off + bytes + 255) & ~(size_t)255; return p; };
  float*  h1   = (float*) carve((size_t)B * C * N * 4);        // fc1 out / f
  __bf16* fnT  = (__bf16*)carve((size_t)B * N * C * 2);        // normalized feats (B,N,C)
  int*    idxb = (int*)   carve((size_t)B * N * KNN * 4);      // knn indices
  __bf16* zT   = (__bf16*)carve((size_t)B * N * 2 * C * 2);    // graph feats (B,N,2C)
  float*  gbuf = (float*) carve((size_t)B * 2 * C * N * 4);    // g conv out
  __bf16* gT   = (__bf16*)carve((size_t)B * N * 2 * C * 2);
  float*  h2   = (float*) carve((size_t)B * C * N * 4);        // fc2 out -> x2
  __bf16* x2T  = (__bf16*)carve((size_t)B * N * C * 2);
  float*  f1o  = (float*) carve((size_t)B * 4 * C * N * 4);    // ffn hidden
  __bf16* f1T  = (__bf16*)carve((size_t)B * N * 4 * C * 2);
  __bf16* xT   = (__bf16*)carve((size_t)B * N * C * 2);
  __bf16* w1b  = (__bf16*)carve((size_t)C * C * 2);
  __bf16* wgb  = (__bf16*)carve((size_t)2 * C * 2 * C * 2);
  __bf16* w2b  = (__bf16*)carve((size_t)C * 2 * C * 2);
  __bf16* wf1b = (__bf16*)carve((size_t)4 * C * C * 2);
  __bf16* wf2b = (__bf16*)carve((size_t)C * 4 * C * 2);

  auto cvt = [&](const float* s, __bf16* d, int n) {
    cvt_f32_bf16<<<(n + 255) / 256, 256, 0, stream>>>(s, d, n);
  };
  cvt(fc1_w, w1b,  C * C);
  cvt(g_w,   wgb,  2 * C * 2 * C);
  cvt(fc2_w, w2b,  C * 2 * C);
  cvt(f1_w,  wf1b, 4 * C * C);
  cvt(f2_w,  wf2b, C * 4 * C);

  int ecnt = B * C * N;
  transpose_cvt<<<(ecnt + 255) / 256, 256, 0, stream>>>(x, xT, C, N, B);

  auto gemm = [&](const __bf16* A, const __bf16* Bt, const float* bias, float* o, int M, int K) {
    gemm_bf16_wmma<<<dim3(N / 64, M / 16, B), 128, 0, stream>>>(A, Bt, bias, o, M, N, K);
  };
  auto bn = [&](float* y, const float* g, const float* bb, const float* res,
                __bf16* oT, int Cc, int relu) {
    bn_fused<<<Cc, 256, 0, stream>>>(y, g, bb, res, oT, Cc, N, B, relu);
  };

  // ---- Grapher ----
  gemm(w1b, xT, fc1_b, h1, C, C);                       // fc1
  bn(h1, bn1_g, bn1_b, nullptr, nullptr, C, 0);         // bn1 (in place)

  normalize_nodes<<<(B * N + 255) / 256, 256, 0, stream>>>(h1, fnT, C, N, B);
  knn_topk<<<dim3(N / 16, B), 128, 0, stream>>>(fnT, idxb, N);
  gather_maxrel<<<(B * C * N + 255) / 256, 256, 0, stream>>>(h1, idxb, zT, C, N, B);

  gemm(wgb, zT, g_b, gbuf, 2 * C, 2 * C);               // graph conv 2C->2C
  bn(gbuf, bng_g, bng_b, nullptr, gT, 2 * C, 1);        // bn + relu, emit bf16T

  gemm(w2b, gT, fc2_b, h2, C, 2 * C);                   // fc2
  bn(h2, bn2_g, bn2_b, x, x2T, C, 0);                   // bn + residual(x) -> x2

  // ---- FFN ----
  gemm(wf1b, x2T, f1_b, f1o, 4 * C, C);                 // f1
  bn(f1o, fbn1_g, fbn1_b, nullptr, f1T, 4 * C, 1);      // bn + relu

  gemm(wf2b, f1T, f2_b, out, C, 4 * C);                 // f2 -> d_out
  bn(out, fbn2_g, fbn2_b, h2, nullptr, C, 0);           // bn + residual(x2), in place
}
